// SwinTransformerBlock_14422500180625
// MI455X (gfx1250) — matrix operand order repaired and checked by reading, not compiled, not run on hardware
//
#include <hip/hip_runtime.h>

// ---------------- types ----------------
typedef __bf16 bf16;
typedef __attribute__((ext_vector_type(16))) __bf16 v16bf;
typedef __attribute__((ext_vector_type(8)))  __bf16 v8bf;
typedef __attribute__((ext_vector_type(8)))  float  v8f;
typedef __attribute__((ext_vector_type(4)))  unsigned u32x4;
typedef __attribute__((ext_vector_type(8)))  unsigned u32x8;

// ---------------- problem constants ----------------
constexpr int cB    = 64;
constexpr int cH    = 56;
constexpr int cW    = 56;
constexpr int cWS   = 7;
constexpr int cSS   = 3;
constexpr int cN    = 49;          // tokens per window (padded to 64 for WMMA)
constexpr int cC    = 192;
constexpr int cDIM  = 128;
constexpr int cAD   = 128;
constexpr int cNH   = 4;
constexpr int cHD   = 32;
constexpr int cHID  = 768;
constexpr int cNWIN = cB * (cH / cWS) * (cW / cWS);   // 4096
constexpr int cTOK  = cB * cH * cW;                   // 200704
constexpr float cSCALE = 0.17677669529663687f;        // 32^-0.5
constexpr float cEPS   = 1e-5f;

// ---------------- helpers ----------------
// Hardware f32 -> bf16 convert (gfx1250 has native BF16 cvt; avoids the
// 5-VALU-op software RNE sequence).
__device__ __forceinline__ bf16 f2bf(float f) { return (bf16)f; }

__device__ __forceinline__ v8f wmma_bf16(v16bf a, v16bf b, v8f c) {
  return __builtin_amdgcn_wmma_f32_16x16x32_bf16(false, a, false, b, (short)0, c,
                                                 false, false);
}

// A operand: 16x32 tile from row-major bf16 (row stride ld elements).
// ISA layout: lanes 0-15 = rows, element halves hold K{0..7,16..23};
// lanes 16-31 hold K{8..15,24..31}.
__device__ __forceinline__ v16bf load_a32(const bf16* __restrict__ tile, int ld) {
  const int lane  = threadIdx.x & 31;
  const int row   = lane & 15;
  const int khalf = (lane >> 4) << 3;           // 0 or 8
  const bf16* p = tile + row * ld + khalf;
  v8bf lo = *(const v8bf*)(p);
  v8bf hi = *(const v8bf*)(p + 16);
  v16bf r;
#pragma unroll
  for (int i = 0; i < 8; ++i) { r[i] = lo[i]; r[8 + i] = hi[i]; }
  return r;
}

// B operand: 32x16 (KxN) tile, read from B^T storage (N rows x K cols, stride ld).
// ISA layout: lanes 0-15 = cols with K{0..7,16..23}, lanes 16-31 = cols with K{8..15,24..31}.
__device__ __forceinline__ v16bf load_bT(const bf16* __restrict__ tileT, int ld) {
  const int lane = threadIdx.x & 31;
  const int col  = lane & 15;
  const int k0   = (lane >> 4) << 3;            // 0 or 8
  const bf16* p = tileT + col * ld + k0;
  v8bf lo = *(const v8bf*)(p);
  v8bf hi = *(const v8bf*)(p + 16);
  v16bf r;
#pragma unroll
  for (int i = 0; i < 8; ++i) { r[i] = lo[i]; r[8 + i] = hi[i]; }
  return r;
}

// ---------------- prep: weight transpose + convert, attention mask ----------------
__global__ void k_transpose_bf16(const float* __restrict__ src, bf16* __restrict__ dst,
                                 int K, int N) {
  int i = blockIdx.x * blockDim.x + threadIdx.x;
  if (i >= K * N) return;
  int k = i / N, n = i % N;
  dst[n * K + k] = f2bf(src[i]);
}

__device__ __forceinline__ int sw_region(int u) {
  return (u < cH - cWS) ? 0 : ((u < cH - cSS) ? 1 : 2);
}

__global__ void k_mask(float* __restrict__ mask) {
  int i = blockIdx.x * blockDim.x + threadIdx.x;
  if (i >= 64 * cN * cN) return;
  int wi = i / (cN * cN);
  int r  = i % (cN * cN);
  int a = r / cN, b = r % cN;                  // query token, key token
  int wh = wi >> 3, ww = wi & 7;
  int ida = 3 * sw_region(wh * cWS + a / cWS) + sw_region(ww * cWS + a % cWS);
  int idb = 3 * sw_region(wh * cWS + b / cWS) + sw_region(ww * cWS + b % cWS);
  mask[i] = (ida != idb) ? -100.0f : 0.0f;
}

// ---------------- LN1 + roll + window partition -> bf16 tiles ----------------
__global__ void k_ln1_window(const float* __restrict__ x, const float* __restrict__ g,
                             const float* __restrict__ be, bf16* __restrict__ hb) {
  int t = blockIdx.x * blockDim.x + threadIdx.x;
  if (t >= cNWIN * 64) return;
  int win = t >> 6, tok = t & 63;
  bf16* dst = hb + (size_t)t * cC;
  if (tok >= cN) {                              // zero padding rows
    for (int c = 0; c < cC; ++c) dst[c] = f2bf(0.0f);
    return;
  }
  int b = win >> 6, wi = win & 63;
  int wh = wi >> 3, ww = wi & 7;
  int ti = tok / cWS, tj = tok % cWS;
  int h0 = wh * cWS + ti + cSS; if (h0 >= cH) h0 -= cH;   // roll(-3)
  int w0 = ww * cWS + tj + cSS; if (w0 >= cW) w0 -= cW;
  const float* row = x + ((size_t)b * (cH * cW) + h0 * cW + w0) * cC;
  float s = 0.f, s2 = 0.f;
  for (int c = 0; c < cC; ++c) { float v = row[c]; s += v; s2 += v * v; }
  float m   = s / cC;
  float var = s2 / cC - m * m;
  float inv = rsqrtf(var + cEPS);
  for (int c = 0; c < cC; ++c)
    dst[c] = f2bf((row[c] - m) * inv * g[c] + be[c]);
}

// ---------------- QKV projections (per window) ----------------
// The 24KB h-window tile is staged global->LDS once per block via the Tensor
// Data Mover (TENSOR_LOAD_TO_LDS, 2-SGPR-group descriptor: 1-D, 3072 x 8B),
// then consumed by 96 WMMA output tiles.
__global__ void __launch_bounds__(256) k_qkv(const bf16* __restrict__ hb,
    const bf16* __restrict__ wqT, const bf16* __restrict__ wkT,
    const bf16* __restrict__ wvT,
    const float* __restrict__ qb, const float* __restrict__ kb,
    const float* __restrict__ vb,
    bf16* __restrict__ qws, bf16* __restrict__ kws, bf16* __restrict__ vtws) {
  __shared__ __align__(16) bf16 hs[64 * cC];    // 24 KB window tile
  const int win  = blockIdx.x;
  const int wave = threadIdx.x >> 5;
  const int lane = threadIdx.x & 31;
  const int col = lane & 15, rbase = (lane >> 4) << 3;

  if (wave == 0) {                              // one TDM issue per workgroup
    const unsigned nelem = (64 * cC * 2) / 8;   // 3072 8-byte elements
    unsigned long long ga = (unsigned long long)(size_t)(hb + (size_t)win * 64 * cC);
    unsigned lds_off = (unsigned)(size_t)(void*)hs;
    u32x4 g0;
    g0[0] = 1u;                                 // D# group0: count=1 (valid, user)
    g0[1] = lds_off;                            // lds_addr
    g0[2] = (unsigned)ga;                       // global_addr[31:0]
    g0[3] = (unsigned)((ga >> 32) & 0x1FFFFFFu) | (2u << 30);  // addr[56:32], type=2
    u32x8 g1;
    g1[0] = 3u << 16;                           // data_size = 8B; no multicast
    g1[1] = (nelem & 0xFFFFu) << 16;            // tensor_dim0[15:0]
    g1[2] = (nelem >> 16) | (1u << 16);         // tensor_dim0[31:16], tensor_dim1=1
    g1[3] = (nelem & 0xFFFFu) << 16;            // tile_dim0 = 3072
    g1[4] = 1u;                                 // tile_dim1 = 1
    g1[5] = nelem;                              // tensor_dim0_stride
    g1[6] = 0u;
    g1[7] = 0u;
    asm volatile("tensor_load_to_lds %0, %1" :: "s"(g0), "s"(g1) : "memory");
    __builtin_amdgcn_s_wait_tensorcnt(0);
  }
  __syncthreads();

  for (int job = wave; job < 96; job += 8) {    // 32 tiles each for q,k,v
    int kind = job >> 5;                        // 0=q 1=k 2=v
    int idx  = job & 31;
    int mt = idx >> 3, nt = idx & 7;
    const bf16* A0 = hs + mt * 16 * cC;
    const bf16* BT; const float* bias; int chunks, ldb;
    if (kind == 0)      { BT = wqT + nt * 16 * cC;   bias = qb; chunks = 6; ldb = cC; }
    else if (kind == 1) { BT = wkT + nt * 16 * cC;   bias = kb; chunks = 6; ldb = cC; }
    else                { BT = wvT + nt * 16 * cDIM; bias = vb; chunks = 4; ldb = cDIM; }
    v8f acc = {};
    for (int c = 0; c < chunks; ++c) {
      v16bf a  = load_a32(A0 + c * 32, cC);
      v16bf bm = load_bT(BT + c * 32, ldb);
      acc = wmma_bf16(a, bm, acc);
    }
    int n = nt * 16 + col;
    float bv = bias[n];
    if (kind == 0) {
      bf16* out = qws + (size_t)win * 64 * cAD;
#pragma unroll
      for (int r = 0; r < 8; ++r)
        out[(mt * 16 + rbase + r) * cAD + n] = f2bf((acc[r] + bv) * cSCALE);
    } else if (kind == 1) {
      bf16* out = kws + (size_t)win * 64 * cAD;
#pragma unroll
      for (int r = 0; r < 8; ++r)
        out[(mt * 16 + rbase + r) * cAD + n] = f2bf(acc[r] + bv);
    } else {                                    // v stored transposed: [d][tok]
      bf16* out = vtws + (size_t)win * cAD * 64;
#pragma unroll
      for (int r = 0; r < 8; ++r)
        out[n * 64 + (mt * 16 + rbase + r)] = f2bf(acc[r] + bv);
    }
  }
}

// ---------------- attention: scores + mask + softmax + PV ----------------
__global__ void __launch_bounds__(256) k_attn(const bf16* __restrict__ qws,
    const bf16* __restrict__ kws, const bf16* __restrict__ vtws,
    const float* __restrict__ mask, bf16* __restrict__ ows) {
  __shared__ float sc[64][65];                       // padded stride: no bank conflicts
  __shared__ __align__(16) bf16 pr[64][64];
  const int win  = blockIdx.x >> 2;
  const int head = blockIdx.x & 3;
  const int wave = threadIdx.x >> 5;
  const int lane = threadIdx.x & 31;
  const int col = lane & 15, rbase = (lane >> 4) << 3;
  const int wi = win & 63;
  const bf16* q = qws + (size_t)win * 64 * cAD + head * cHD;
  const bf16* k = kws + (size_t)win * 64 * cAD + head * cHD;

  // phase 1: S = q k^T (K = HD = 32, one wmma per 16x16 tile)
  for (int t = wave; t < 16; t += 8) {
    int mt = t >> 2, nt = t & 3;
    v16bf a  = load_a32(q + mt * 16 * cAD, cAD);
    v16bf bm = load_bT(k + nt * 16 * cAD, cAD);
    v8f acc = {};
    acc = wmma_bf16(a, bm, acc);
#pragma unroll
    for (int r = 0; r < 8; ++r) {
      int m = mt * 16 + rbase + r;
      int n = nt * 16 + col;
      float v = acc[r];
      if (n >= cN)      v = -1e30f;                              // padded keys
      else if (m < cN)  v += mask[(size_t)wi * cN * cN + m * cN + n];
      sc[m][n] = v;
    }
  }
  __syncthreads();

  // phase 2: row softmax (three passes, low register pressure)
  if (threadIdx.x < 64) {
    int row = threadIdx.x;
    float mx = -1e30f;
    for (int n = 0; n < 64; ++n) mx = fmaxf(mx, sc[row][n]);
    float s = 0.f;
    for (int n = 0; n < 64; ++n) s += expf(sc[row][n] - mx);
    float inv = 1.0f / s;
    for (int n = 0; n < 64; ++n) pr[row][n] = f2bf(expf(sc[row][n] - mx) * inv);
  }
  __syncthreads();

  // phase 3: O = P V (K = 64 tokens = 2 chunks), one tile per wave
  {
    int mt = wave >> 1, nt = wave & 1;
    const bf16* vt = vtws + (size_t)win * cAD * 64 + (head * cHD + nt * 16) * 64;
    v8f acc = {};
#pragma unroll
    for (int c = 0; c < 2; ++c) {
      v16bf a  = load_a32(&pr[mt * 16][c * 32], 64);
      v16bf bm = load_bT(vt + c * 32, 64);
      acc = wmma_bf16(a, bm, acc);
    }
    bf16* out = ows + (size_t)win * 64 * cAD;
#pragma unroll
    for (int r = 0; r < 8; ++r)
      out[(mt * 16 + rbase + r) * cAD + head * cHD + nt * 16 + col] = f2bf(acc[r]);
  }
}

// ---------------- output proj + window reverse + roll back + residual ----------------
__global__ void __launch_bounds__(256) k_proj(const bf16* __restrict__ ows,
    const bf16* __restrict__ wpT, const float* __restrict__ pb,
    const float* __restrict__ x, float* __restrict__ x2) {
  const int win  = blockIdx.x;
  const int wave = threadIdx.x >> 5;
  const int lane = threadIdx.x & 31;
  const int col = lane & 15, rbase = (lane >> 4) << 3;
  const int b = win >> 6, wi = win & 63;
  const int wh = wi >> 3, ww = wi & 7;
  const bf16* o = ows + (size_t)win * 64 * cAD;
  for (int job = wave; job < 48; job += 8) {    // 4 M-tiles x 12 N-tiles
    int mt = job / 12, nt = job % 12;
    v8f acc = {};
    for (int c = 0; c < 4; ++c) {
      v16bf a  = load_a32(o + mt * 16 * cAD + c * 32, cAD);
      v16bf bm = load_bT(wpT + nt * 16 * cAD + c * 32, cAD);
      acc = wmma_bf16(a, bm, acc);
    }
    int n = nt * 16 + col;
    float bv = pb[n];
#pragma unroll
    for (int r = 0; r < 8; ++r) {
      int m = mt * 16 + rbase + r;
      if (m < cN) {
        int ti = m / cWS, tj = m % cWS;
        int h0 = wh * cWS + ti + cSS; if (h0 >= cH) h0 -= cH;
        int w0 = ww * cWS + tj + cSS; if (w0 >= cW) w0 -= cW;
        size_t pos = ((size_t)b * (cH * cW) + h0 * cW + w0) * cC + n;
        x2[pos] = x[pos] + acc[r] + bv;
      }
    }
  }
}

// ---------------- LN2 -> bf16 ----------------
__global__ void k_ln2(const float* __restrict__ x2, const float* __restrict__ g,
                      const float* __restrict__ be, bf16* __restrict__ xn) {
  int t = blockIdx.x * blockDim.x + threadIdx.x;
  if (t >= cTOK) return;
  const float* row = x2 + (size_t)t * cC;
  float s = 0.f, s2 = 0.f;
  for (int c = 0; c < cC; ++c) { float v = row[c]; s += v; s2 += v * v; }
  float m   = s / cC;
  float var = s2 / cC - m * m;
  float inv = rsqrtf(var + cEPS);
  bf16* dst = xn + (size_t)t * cC;
  for (int c = 0; c < cC; ++c)
    dst[c] = f2bf((row[c] - m) * inv * g[c] + be[c]);
}

// ---------------- FC1 + exact GELU ----------------
__global__ void __launch_bounds__(256) k_fc1(const bf16* __restrict__ xn,
    const bf16* __restrict__ w1T, const float* __restrict__ b1,
    bf16* __restrict__ h2) {
  const int mblk = blockIdx.x / 6;              // 64-token block
  const int nblk = blockIdx.x % 6;              // 128-col slab of HID
  const int wave = threadIdx.x >> 5;
  const int lane = threadIdx.x & 31;
  const int col = lane & 15, rbase = (lane >> 4) << 3;
  const bf16* A = xn + (size_t)mblk * 64 * cC;
  for (int job = wave; job < 32; job += 8) {
    int mt = job >> 3, nt = job & 7;
    int nb = nblk * 128 + nt * 16;
    const bf16* BT = w1T + (size_t)nb * cC;
    v8f acc = {};
    for (int c = 0; c < 6; ++c) {
      if (c + 1 < 6) __builtin_prefetch(BT + (c + 1) * 32, 0, 1);  // global_prefetch_b8
      v16bf a  = load_a32(A + mt * 16 * cC + c * 32, cC);
      v16bf bm = load_bT(BT + c * 32, cC);
      acc = wmma_bf16(a, bm, acc);
    }
    int n = nb + col;
    float bv = b1[n];
#pragma unroll
    for (int r = 0; r < 8; ++r) {
      size_t m = (size_t)mblk * 64 + mt * 16 + rbase + r;
      float z = acc[r] + bv;
      float gel = 0.5f * z * (1.0f + erff(z * 0.70710678118654752f));
      h2[m * cHID + n] = f2bf(gel);
    }
  }
}

// ---------------- FC2 + residual -> out ----------------
__global__ void __launch_bounds__(256) k_fc2(const bf16* __restrict__ h2,
    const bf16* __restrict__ w2T, const float* __restrict__ b2,
    const float* __restrict__ x2, float* __restrict__ out) {
  const int mblk = blockIdx.x;
  const int wave = threadIdx.x >> 5;
  const int lane = threadIdx.x & 31;
  const int col = lane & 15, rbase = (lane >> 4) << 3;
  const bf16* A = h2 + (size_t)mblk * 64 * cHID;
  for (int job = wave; job < 48; job += 8) {    // 4 M-tiles x 12 N-tiles, K=768
    int mt = job / 12, nt = job % 12;
    const bf16* BT = w2T + (size_t)(nt * 16) * cHID;
    v8f acc = {};
    for (int c = 0; c < 24; ++c) {
      if (c + 1 < 24) __builtin_prefetch(BT + (c + 1) * 32, 0, 1);
      v16bf a  = load_a32(A + mt * 16 * cHID + c * 32, cHID);
      v16bf bm = load_bT(BT + c * 32, cHID);
      acc = wmma_bf16(a, bm, acc);
    }
    int n = nt * 16 + col;
    float bv = b2[n];
#pragma unroll
    for (int r = 0; r < 8; ++r) {
      size_t m = (size_t)mblk * 64 + mt * 16 + rbase + r;
      out[m * cC + n] = x2[m * cC + n] + acc[r] + bv;
    }
  }
}

// ---------------- launcher ----------------
extern "C" void kernel_launch(void* const* d_in, const int* in_sizes, int n_in,
                              void* d_out, int out_size, void* d_ws, size_t ws_size,
                              hipStream_t stream) {
  (void)in_sizes; (void)n_in; (void)out_size; (void)ws_size;
  const float* x   = (const float*)d_in[0];
  const float* g1  = (const float*)d_in[1];
  const float* be1 = (const float*)d_in[2];
  const float* qw  = (const float*)d_in[3];
  const float* qb  = (const float*)d_in[4];
  const float* kw  = (const float*)d_in[5];
  const float* kb  = (const float*)d_in[6];
  const float* vw  = (const float*)d_in[7];
  const float* vb  = (const float*)d_in[8];
  const float* pw  = (const float*)d_in[9];
  const float* pb  = (const float*)d_in[10];
  const float* g2  = (const float*)d_in[11];
  const float* be2 = (const float*)d_in[12];
  const float* w1  = (const float*)d_in[13];
  const float* b1  = (const float*)d_in[14];
  const float* w2  = (const float*)d_in[15];
  const float* b2  = (const float*)d_in[16];
  float* out = (float*)d_out;

  char* ws = (char*)d_ws;
  size_t off = 0;
  auto take = [&](size_t bytes) -> char* {
    char* p = ws + off;
    off += (bytes + 255) & ~(size_t)255;
    return p;
  };
  bf16* wqT  = (bf16*)take((size_t)cAD * cC  * 2);   // (128 x 192)
  bf16* wkT  = (bf16*)take((size_t)cAD * cC  * 2);
  bf16* wvT  = (bf16*)take((size_t)cAD * cDIM * 2);  // (128 x 128)
  bf16* wpT  = (bf16*)take((size_t)cC  * cAD * 2);   // (192 x 128)
  bf16* w1T  = (bf16*)take((size_t)cHID * cC * 2);   // (768 x 192)
  bf16* w2T  = (bf16*)take((size_t)cC * cHID * 2);   // (192 x 768)
  float* mk  = (float*)take((size_t)64 * cN * cN * 4);
  bf16* hb   = (bf16*)take((size_t)cNWIN * 64 * cC * 2);
  bf16* qws  = (bf16*)take((size_t)cNWIN * 64 * cAD * 2);
  bf16* kws  = (bf16*)take((size_t)cNWIN * 64 * cAD * 2);
  bf16* vtws = (bf16*)take((size_t)cNWIN * cAD * 64 * 2);
  bf16* ows  = (bf16*)take((size_t)cNWIN * 64 * cAD * 2);
  float* x2  = (float*)take((size_t)cTOK * cC * 4);
  bf16* xn   = (bf16*)take((size_t)cTOK * cC * 2);
  bf16* h2   = (bf16*)take((size_t)cTOK * cHID * 2);

  // prep
  k_transpose_bf16<<<(cC * cAD + 255) / 256, 256, 0, stream>>>(qw, wqT, cC, cAD);
  k_transpose_bf16<<<(cC * cAD + 255) / 256, 256, 0, stream>>>(kw, wkT, cC, cAD);
  k_transpose_bf16<<<(cDIM * cAD + 255) / 256, 256, 0, stream>>>(vw, wvT, cDIM, cAD);
  k_transpose_bf16<<<(cAD * cC + 255) / 256, 256, 0, stream>>>(pw, wpT, cAD, cC);
  k_transpose_bf16<<<(cC * cHID + 255) / 256, 256, 0, stream>>>(w1, w1T, cC, cHID);
  k_transpose_bf16<<<(cHID * cC + 255) / 256, 256, 0, stream>>>(w2, w2T, cHID, cC);
  k_mask<<<(64 * cN * cN + 255) / 256, 256, 0, stream>>>(mk);

  // attention path
  k_ln1_window<<<cNWIN * 64 / 256, 256, 0, stream>>>(x, g1, be1, hb);
  k_qkv<<<cNWIN, 256, 0, stream>>>(hb, wqT, wkT, wvT, qb, kb, vb, qws, kws, vtws);
  k_attn<<<cNWIN * cNH, 256, 0, stream>>>(qws, kws, vtws, mk, ows);
  k_proj<<<cNWIN, 256, 0, stream>>>(ows, wpT, pb, x, x2);

  // MLP path
  k_ln2<<<(cTOK + 255) / 256, 256, 0, stream>>>(x2, g2, be2, xn);
  k_fc1<<<(cTOK / 64) * 6, 256, 0, stream>>>(xn, w1T, b1, h2);
  k_fc2<<<cTOK / 64, 256, 0, stream>>>(h2, w2T, b2, x2, out);
}
